// LogEuclideanResidual_22643067584705
// MI455X (gfx1250) — matrix operand order repaired and checked by reading, not compiled, not run on hardware
//
#include <hip/hip_runtime.h>
#include <hip/hip_bf16.h>
#include <math.h>

// ---------------------------------------------------------------------------
// Z = expm( logm(X) + logm(Y) ) for 16384 SPD 32x32 f32 matrices.
// One wave32 per matrix. All matrix functions are built from 32x32 matmuls,
// each matmul = 4x v_wmma_f32_16x16x32_bf16 (f32 accumulate).
// All LDS buffers are FIXED addresses (no pointer phis) so address-space
// inference produces ds_* ops, not flat_*.
// ---------------------------------------------------------------------------

typedef __attribute__((ext_vector_type(16))) __bf16 v16bf;
typedef __attribute__((ext_vector_type(8)))  float  v8f;

#define LDSTRIDE 33              // pad rows: 33 is coprime with 64 banks
#define MATSZ    (32 * LDSTRIDE) // floats per LDS matrix buffer
#define NBUF     6               // Lb + 5 fixed scratch matrices per wave

union BFrag { v16bf v; __bf16 e[16]; };

// D = A * B  (32x32 f32 matrices in LDS, row stride LDSTRIDE). One wave,
// EXEC all ones. Emits 4x v_wmma_f32_16x16x32_bf16.
__device__ __forceinline__ void wave_mm(float* D, const float* A, const float* B,
                                        int lane) {
  const int hi = (lane >> 4) & 1;   // upper half-wave?
  const int lr = lane & 15;
  BFrag a0, a1, b0, b1;
#pragma unroll
  for (int i = 0; i < 16; ++i) {
    // A fragment (16x32, 16-bit): lanes 0-15 K=0..15, lanes 16-31 K=16..31
    const int ka = i + (hi ? 16 : 0);
    a0.e[i] = (__bf16)A[lr * LDSTRIDE + ka];
    a1.e[i] = (__bf16)A[(16 + lr) * LDSTRIDE + ka];
    // B fragment (32x16, 16-bit): lanes 0-15 K=0..15, lanes 16-31 K=16..31
    const int kb = i + (hi ? 16 : 0);
    b0.e[i] = (__bf16)B[kb * LDSTRIDE + lr];
    b1.e[i] = (__bf16)B[kb * LDSTRIDE + 16 + lr];
  }
  v8f c00 = {}, c01 = {}, c10 = {}, c11 = {};
  c00 = __builtin_amdgcn_wmma_f32_16x16x32_bf16(false, a0.v, false, b0.v,
                                                (short)0, c00, false, false);
  c01 = __builtin_amdgcn_wmma_f32_16x16x32_bf16(false, a0.v, false, b1.v,
                                                (short)0, c01, false, false);
  c10 = __builtin_amdgcn_wmma_f32_16x16x32_bf16(false, a1.v, false, b0.v,
                                                (short)0, c10, false, false);
  c11 = __builtin_amdgcn_wmma_f32_16x16x32_bf16(false, a1.v, false, b1.v,
                                                (short)0, c11, false, false);
#pragma unroll
  for (int r = 0; r < 8; ++r) {
    const int m = r + (hi ? 8 : 0);      // C/D layout: M = r + 8*(lane>=16)
    D[m * LDSTRIDE + lr]             = c00[r];
    D[m * LDSTRIDE + 16 + lr]        = c01[r];
    D[(16 + m) * LDSTRIDE + lr]      = c10[r];
    D[(16 + m) * LDSTRIDE + 16 + lr] = c11[r];
  }
}

__device__ __forceinline__ float wave_frob(const float* M, int lane) {
  float s = 0.f;
#pragma unroll
  for (int i = 0; i < 32; ++i) {
    float v = M[lane * LDSTRIDE + i];
    s += v * v;
  }
#pragma unroll
  for (int m = 16; m >= 1; m >>= 1) s += __shfl_xor(s, m, 32);
  return sqrtf(s);
}

// dst = beta*src + alpha*I  (dst may alias src); lane owns row `lane`.
__device__ __forceinline__ void ew_axpbi(float* dst, const float* src,
                                         float beta, float alpha, int lane) {
#pragma unroll
  for (int c = 0; c < 32; ++c)
    dst[lane * LDSTRIDE + c] =
        beta * src[lane * LDSTRIDE + c] + ((c == lane) ? alpha : 0.f);
}

__device__ __forceinline__ void ew_seti(float* dst, float alpha, int lane) {
#pragma unroll
  for (int c = 0; c < 32; ++c)
    dst[lane * LDSTRIDE + c] = (c == lane) ? alpha : 0.f;
}

// Lb = (acc ? Lb : 0) + beta*src + alpha*I
__device__ __forceinline__ void ew_acc(float* Lb, const float* src, float beta,
                                       float alpha, bool acc, int lane) {
#pragma unroll
  for (int c = 0; c < 32; ++c) {
    float v = beta * src[lane * LDSTRIDE + c] + ((c == lane) ? alpha : 0.f);
    float o = acc ? Lb[lane * LDSTRIDE + c] : 0.f;
    Lb[lane * LDSTRIDE + c] = o + v;
  }
}

// Lb (+)= logm(g). Inverse scaling-and-squaring with Newton-Schulz sqrts.
// Buffers y,z,t,u,w are fixed LDS addresses; iterations are paired so the
// ping-pong never swaps pointers.
__device__ __forceinline__ void
wave_logm_acc(const float* __restrict__ g, float* Lb, float* y, float* z,
              float* t, float* u, float* w, bool acc, int lane) {
  // coalesced load into y
#pragma unroll
  for (int i = 0; i < 32; ++i) y[i * LDSTRIDE + lane] = g[i * 32 + lane];

  const float c    = wave_frob(y, lane) * 1.000001f + 1e-30f;
  const float invc = 1.f / c;
#pragma unroll
  for (int i = 0; i < 32; ++i) y[lane * LDSTRIDE + i] *= invc;  // spectrum (0,1]

  // 4 coupled Newton-Schulz sqrt levels: y -> A^(1/2), z -> A^(-1/2)
  ew_seti(z, 1.f, lane);
#pragma unroll 1
  for (int lev = 0; lev < 4; ++lev) {
    const int pairs = (lev == 0) ? 6 : ((lev == 1) ? 4 : ((lev == 2) ? 3 : 2));
#pragma unroll 1
    for (int p = 0; p < pairs; ++p) {
      // iter A: (y,z) -> (u,w)
      wave_mm(t, z, y, lane);
      ew_axpbi(t, t, -0.5f, 1.5f, lane);   // t = 1.5I - 0.5 z*y
      wave_mm(u, y, t, lane);              // u = new y
      wave_mm(w, t, z, lane);              // w = new z
      // iter B: (u,w) -> (y,z)
      wave_mm(t, w, u, lane);
      ew_axpbi(t, t, -0.5f, 1.5f, lane);
      wave_mm(y, u, t, lane);
      wave_mm(z, t, w, lane);
    }
    if (lev < 3) ew_seti(z, 1.f, lane);    // next level: A := y, Z0 = I
  }

  // M = y = (X/c)^(1/16), ||I-M|| <= ~0.25.  Gregory series, 13 terms:
  // log(M) = -sum_{j=1..13} E^j / j,  E = I - M   (Horner, 12 matmuls)
  ew_axpbi(z, y, -1.f, 1.f, lane);         // z = E
  ew_seti(t, 1.f / 13.f, lane);
#pragma unroll 1
  for (int j = 12; j >= 2; j -= 2) {
    wave_mm(u, z, t, lane);                              // u = E*S
    ew_axpbi(u, u, 1.f, 1.f / (float)j, lane);           // S = E*S + (1/j)I
    wave_mm(t, z, u, lane);
    ew_axpbi(t, t, 1.f, 1.f / (float)(j - 1), lane);
  }
  // log(X) = 16 * log(M) + log(c) I = -16 * S + log(c) I
  ew_acc(Lb, t, -16.f, logf(c), acc, lane);
}

__global__ __launch_bounds__(128) void
LogEuclideanResidual_22643067584705_kernel(const float* __restrict__ X,
                                           const float* __restrict__ Y,
                                           float* __restrict__ Out, int batch) {
  extern __shared__ float smem[];
  const int lane = threadIdx.x & 31;
  const int wid  = threadIdx.x >> 5;
  const int mat  = blockIdx.x * (blockDim.x >> 5) + wid;
  if (mat >= batch) return;  // wave-uniform; EXEC stays all-1s inside

  float* Lb = &smem[wid * (NBUF * MATSZ) + 0 * MATSZ];
  float* b0 = &smem[wid * (NBUF * MATSZ) + 1 * MATSZ];
  float* b1 = &smem[wid * (NBUF * MATSZ) + 2 * MATSZ];
  float* b2 = &smem[wid * (NBUF * MATSZ) + 3 * MATSZ];
  float* b3 = &smem[wid * (NBUF * MATSZ) + 4 * MATSZ];
  float* b4 = &smem[wid * (NBUF * MATSZ) + 5 * MATSZ];

  const float* xs = X + (size_t)mat * 1024;
  const float* ys = Y + (size_t)mat * 1024;

  wave_logm_acc(xs, Lb, b0, b1, b2, b3, b4, /*acc=*/false, lane);
  wave_logm_acc(ys, Lb, b0, b1, b2, b3, b4, /*acc=*/true, lane);

  // expm(Lb): scaling-and-squaring, s=4, 10-term Taylor (Horner, fixed bufs)
  ew_axpbi(b0, Lb, 1.f / 16.f, 0.f, lane);  // P = S / 2^4   (in b0)
  ew_seti(b1, 1.f, lane);                   // R = I         (t = b1, u = b2)
#pragma unroll 1
  for (int j = 10; j >= 2; j -= 2) {
    wave_mm(b2, b0, b1, lane);                             // u = P*R
    ew_axpbi(b2, b2, 1.f / (float)j, 1.f, lane);           // R = I + (P*R)/j
    wave_mm(b1, b0, b2, lane);
    ew_axpbi(b1, b1, 1.f / (float)(j - 1), 1.f, lane);
  }
  // R = R^(2^4): 4 squarings, ends in b1
  wave_mm(b2, b1, b1, lane);
  wave_mm(b1, b2, b2, lane);
  wave_mm(b2, b1, b1, lane);
  wave_mm(b1, b2, b2, lane);

  float* out = Out + (size_t)mat * 1024;
#pragma unroll
  for (int i = 0; i < 32; ++i) out[i * 32 + lane] = b1[i * LDSTRIDE + lane];
}

extern "C" void kernel_launch(void* const* d_in, const int* in_sizes, int n_in,
                              void* d_out, int out_size, void* d_ws, size_t ws_size,
                              hipStream_t stream) {
  const float* X = (const float*)d_in[0];
  const float* Y = (const float*)d_in[1];
  float* O = (float*)d_out;
  const int batch = in_sizes[0] / 1024;          // 16384 matrices of 32x32

  const int wavesPerBlock = 4;                   // 128 threads
  const int blocks = (batch + wavesPerBlock - 1) / wavesPerBlock;
  const size_t smemBytes =
      (size_t)wavesPerBlock * NBUF * MATSZ * sizeof(float);  // ~99 KB / block

  hipLaunchKernelGGL(LogEuclideanResidual_22643067584705_kernel, dim3(blocks),
                     dim3(wavesPerBlock * 32), smemBytes, stream, X, Y, O,
                     batch);
}
